// MultiHeadedAttention_33294586478773
// MI455X (gfx1250) — compile-verified, matrix operand order checked by reading, not run on hardware
//
#include <hip/hip_runtime.h>

// -------------------- problem constants --------------------
#define B_   8
#define S_   1024
#define H_   8
#define DM_  1024
#define DC_  768
#define DP_  256
#define DKC_ 48
#define DKP_ 16
#define DH_  64     // per-head dim (48 content + 16 position)
#define ND_  33     // distance buckets (2*16+1)

// -------------------- types / bf16 helpers --------------------
typedef __attribute__((ext_vector_type(16))) __bf16        bf16x16;
typedef __attribute__((ext_vector_type(8)))  float         f32x8;
typedef __attribute__((ext_vector_type(4)))  unsigned int  u32x4;

// WMMA operand: 16 bf16 per lane = two aligned 128-bit chunks.
union BV16 { bf16x16 v; unsigned short s[16]; u32x4 q[2]; };

__device__ __forceinline__ unsigned short f2bf(float f) {
  union { float f; unsigned int u; } x; x.f = f;
  unsigned int r = 0x7FFFu + ((x.u >> 16) & 1u);   // round-to-nearest-even
  return (unsigned short)((x.u + r) >> 16);
}
__device__ __forceinline__ float bf2f(unsigned short h) {
  union { unsigned int u; float f; } x; x.u = ((unsigned int)h) << 16;
  return x.f;
}
__device__ __forceinline__ void load_op(BV16& d, const unsigned short* p) {
  d.q[0] = *(const u32x4*)(p);        // lane elements e0..7   (K = base .. base+7)
  d.q[1] = *(const u32x4*)(p + 16);   // lane elements e8..15  (K = base+16 .. +23)
}

// -------------------- weight convert + transpose: Wt[n*K+k] = bf16(W[k*N+n]) ---
__global__ void convt_kernel(const float* __restrict__ W,
                             unsigned short* __restrict__ Wt, int K, int N) {
  int idx = blockIdx.x * blockDim.x + threadIdx.x;
  if (idx >= K * N) return;
  int k = idx / N, n = idx % N;
  Wt[(size_t)n * K + k] = f2bf(W[idx]);
}

// -------------------- input convert: Y[r*cols+c] = bf16(X[r*1024+c]) ----------
__global__ void cvt_input_kernel(const float* __restrict__ X,
                                 unsigned short* __restrict__ Y, int cols) {
  int idx = blockIdx.x * blockDim.x + threadIdx.x;
  if (idx >= B_ * S_ * cols) return;
  int r = idx / cols, c = idx % cols;
  Y[idx] = f2bf(X[(size_t)r * DM_ + c]);
}

// -------------------- all-bf16 WMMA GEMM --------------------
// Out[M,N] = scale * (A[M,K] @ Wt^T + bias), A bf16 row stride lda, Wt bf16 [N][K].
// One wave computes a 16x64 strip: A operand reused across 4 WMMAs per k-step.
template <bool OUT_F32>
__global__ void wmma_gemm_bf16(const unsigned short* __restrict__ A, int lda,
                               const unsigned short* __restrict__ Wt,
                               const float* __restrict__ bias,
                               void* __restrict__ outp, int ldo,
                               int Mt, int Nt4, int K, float scale) {
  int wave = threadIdx.x >> 5;
  int lane = threadIdx.x & 31;
  int tile = blockIdx.x * (blockDim.x >> 5) + wave;
  if (tile >= Mt * Nt4) return;
  int tm = tile % Mt, tn = tile / Mt;
  int hlf = lane >> 4, lm = lane & 15;

  f32x8 acc[4] = {};
  const unsigned short* arow = A + (size_t)(tm * 16 + lm) * lda + hlf * 8;
  const unsigned short* wbase = Wt + (size_t)(tn * 64 + lm) * K + hlf * 8;
  for (int k0 = 0; k0 < K; k0 += 32) {
    BV16 a;
    load_op(a, arow + k0);
#pragma unroll
    for (int j = 0; j < 4; ++j) {
      BV16 b;
      load_op(b, wbase + (size_t)(j * 16) * K + k0);
      acc[j] = __builtin_amdgcn_wmma_f32_16x16x32_bf16(false, a.v, false, b.v,
                                                       (short)0, acc[j], false, false);
    }
  }
#pragma unroll
  for (int j = 0; j < 4; ++j) {
    int col = tn * 64 + j * 16 + lm;
    float bv = bias ? bias[col] : 0.0f;
#pragma unroll
    for (int i = 0; i < 8; ++i) {
      int row = tm * 16 + hlf * 8 + i;
      float val = (acc[j][i] + bv) * scale;
      if (OUT_F32) ((float*)outp)[(size_t)row * ldo + col] = val;
      else ((unsigned short*)outp)[(size_t)row * ldo + col] = f2bf(val);
    }
  }
}

// -------- reorder [B*S, H*48]+[B*S, H*16] -> [B,H,S,64] (rows) or [B,H,64,S] (T)
template <bool TRANSPOSED>
__global__ void build_heads_kernel(const unsigned short* __restrict__ Xc,
                                   const unsigned short* __restrict__ Xp,
                                   unsigned short* __restrict__ out) {
  int idx = blockIdx.x * blockDim.x + threadIdx.x;
  if (idx >= B_ * H_ * S_ * DH_) return;
  int d = idx & (DH_ - 1);
  int s = (idx >> 6) & (S_ - 1);
  int h = (idx >> 16) & (H_ - 1);
  int b = idx >> 19;
  size_t row = (size_t)b * S_ + s;
  unsigned short v = (d < DKC_) ? Xc[row * (H_ * DKC_) + h * DKC_ + d]
                                : Xp[row * (H_ * DKP_) + h * DKP_ + (d - DKC_)];
  size_t bh = (size_t)b * H_ + h;
  if (TRANSPOSED) out[(bh * DH_ + d) * S_ + s] = v;   // [B,H,64,S]
  else            out[(bh * S_ + s) * DH_ + d] = v;   // [B,H,S,64]
}

// -------------------- rel-key logits R[bh, q, d] = q . ek[d] -------------------
__global__ void rel_logits_kernel(const unsigned short* __restrict__ Qb,
                                  const float* __restrict__ ek_c,
                                  const float* __restrict__ ek_p,
                                  float* __restrict__ R) {
  int idx = blockIdx.x * blockDim.x + threadIdx.x;
  if (idx >= B_ * H_ * S_ * ND_) return;
  int d = idx % ND_;
  int q = (idx / ND_) % S_;
  int bh = idx / (ND_ * S_);
  const unsigned short* qp = Qb + ((size_t)bh * S_ + q) * DH_;
  float acc = 0.0f;
#pragma unroll 8
  for (int k = 0; k < DKC_; ++k) acc += bf2f(qp[k]) * ek_c[d * DKC_ + k];
#pragma unroll
  for (int k = 0; k < DKP_; ++k) acc += bf2f(qp[DKC_ + k]) * ek_p[d * DKP_ + k];
  R[idx] = acc;
}

// -------------------- flash attention with rel-pos buckets ---------------------
// 1 wave / block; 16 query rows per block. grid = B*H*(S/16).
// Qb/Kb: [B,H,S,64] bf16; Vt: [B,H,64,S] bf16 (so P.V B-operand loads are b128).
__global__ __launch_bounds__(32) void attn_kernel(
    const unsigned short* __restrict__ Qb, const unsigned short* __restrict__ Kb,
    const unsigned short* __restrict__ Vt, const float* __restrict__ R,
    const unsigned char* __restrict__ mask,   // bool [B,S,S]
    const float* __restrict__ ev_c, const float* __restrict__ ev_p,
    unsigned short* __restrict__ ctxc, unsigned short* __restrict__ ctxp) {
  __shared__ float bucket[16 * ND_];                 // unnormalized rel-V mass
  __shared__ float lfac[16];                         // per-row rescale factor
  __shared__ __align__(16) unsigned short pbuf[16 * 32];  // staged P tile (bf16)

  int lane = threadIdx.x;
  int hlf = lane >> 4, lm = lane & 15;
  int qt = blockIdx.x & 63;
  int bh = blockIdx.x >> 6;
  int b = bh >> 3, h = bh & 7;

  for (int i = lane; i < 16 * ND_; i += 32) bucket[i] = 0.0f;

  // q A-operands: dims 0..31 (aq0), 32..63 (aq1)
  BV16 aq0, aq1;
  {
    const unsigned short* qp = Qb + ((size_t)bh * S_ + qt * 16 + lm) * DH_ + hlf * 8;
    load_op(aq0, qp);
    load_op(aq1, qp + 32);
  }

  float m_i[8], l_i[8];
  f32x8 cacc[4] = {};
#pragma unroll
  for (int i = 0; i < 8; ++i) { m_i[i] = -3.0e38f; l_i[i] = 0.0f; }
  __syncthreads();

  for (int kb = 0; kb < S_; kb += 32) {
    float p[2][8], tmax[8];
#pragma unroll
    for (int i = 0; i < 8; ++i) tmax[i] = -3.0e38f;

#pragma unroll
    for (int sub = 0; sub < 2; ++sub) {
      BV16 bk0, bk1;
      const unsigned short* kp =
          Kb + ((size_t)bh * S_ + kb + sub * 16 + lm) * DH_ + hlf * 8;
      load_op(bk0, kp);
      load_op(bk1, kp + 32);
      f32x8 s = {};
      s = __builtin_amdgcn_wmma_f32_16x16x32_bf16(false, aq0.v, false, bk0.v,
                                                  (short)0, s, false, false);
      s = __builtin_amdgcn_wmma_f32_16x16x32_bf16(false, aq1.v, false, bk1.v,
                                                  (short)0, s, false, false);
#pragma unroll
      for (int i = 0; i < 8; ++i) {
        int qg = qt * 16 + hlf * 8 + i;
        int kg = kb + sub * 16 + lm;
        int dr = kg - qg; dr = dr < -16 ? -16 : (dr > 16 ? 16 : dr);
        float sv = s[i] + R[((size_t)bh * S_ + qg) * ND_ + dr + 16];
        if (mask[((size_t)b * S_ + qg) * S_ + kg]) sv = -1.0e30f;
        p[sub][i] = sv;
        tmax[i] = fmaxf(tmax[i], sv);
      }
    }
    // row max / sum across the 16 lanes holding one row
#pragma unroll
    for (int off = 1; off < 16; off <<= 1)
#pragma unroll
      for (int i = 0; i < 8; ++i)
        tmax[i] = fmaxf(tmax[i], __shfl_xor(tmax[i], off, 32));

    float f[8], rsum[8];
#pragma unroll
    for (int i = 0; i < 8; ++i) {
      float nm = fmaxf(m_i[i], tmax[i]);
      f[i] = __expf(m_i[i] - nm);
      m_i[i] = nm;
      p[0][i] = __expf(p[0][i] - nm);
      p[1][i] = __expf(p[1][i] - nm);
      rsum[i] = p[0][i] + p[1][i];
    }
#pragma unroll
    for (int off = 1; off < 16; off <<= 1)
#pragma unroll
      for (int i = 0; i < 8; ++i) rsum[i] += __shfl_xor(rsum[i], off, 32);
#pragma unroll
    for (int i = 0; i < 8; ++i) l_i[i] = l_i[i] * f[i] + rsum[i];
#pragma unroll
    for (int nt = 0; nt < 4; ++nt)
#pragma unroll
      for (int i = 0; i < 8; ++i) cacc[nt][i] *= f[i];

    // rescale distance buckets by per-row factor
    if (lm == 0)
#pragma unroll
      for (int i = 0; i < 8; ++i) lfac[hlf * 8 + i] = f[i];
    __syncthreads();
    for (int idx = lane; idx < 16 * ND_; idx += 32) bucket[idx] *= lfac[idx / ND_];
    __syncthreads();

    // accumulate buckets + stage P (bf16) for the P.V WMMA
#pragma unroll
    for (int sub = 0; sub < 2; ++sub)
#pragma unroll
      for (int i = 0; i < 8; ++i) {
        int qg = qt * 16 + hlf * 8 + i;
        int kg = kb + sub * 16 + lm;
        int dr = kg - qg; dr = dr < -16 ? -16 : (dr > 16 ? 16 : dr);
        atomicAdd(&bucket[(hlf * 8 + i) * ND_ + dr + 16], p[sub][i]);
        pbuf[(hlf * 8 + i) * 32 + sub * 16 + lm] = f2bf(p[sub][i]);
      }
    __syncthreads();

    BV16 ap;
    load_op(ap, &pbuf[lm * 32 + hlf * 8]);   // ds_load_b128 x2
#pragma unroll
    for (int nt = 0; nt < 4; ++nt) {
      BV16 bvv;
      // Vt row = dim (nt*16+lm); contiguous in k -> two b128 loads
      load_op(bvv, Vt + ((size_t)bh * DH_ + nt * 16 + lm) * S_ + kb + hlf * 8);
      cacc[nt] = __builtin_amdgcn_wmma_f32_16x16x32_bf16(
          false, ap.v, false, bvv.v, (short)0, cacc[nt], false, false);
    }
    __syncthreads();
  }

  // epilogue: add bucket @ ev, normalize, scatter to content/position streams
#pragma unroll
  for (int nt = 0; nt < 4; ++nt) {
    int dim = nt * 16 + lm;
#pragma unroll
    for (int i = 0; i < 8; ++i) {
      int r = hlf * 8 + i;
      float rel = 0.0f;
      for (int d = 0; d < ND_; ++d) {
        float e = (dim < DKC_) ? ev_c[d * DKC_ + dim] : ev_p[d * DKP_ + (dim - DKC_)];
        rel += bucket[r * ND_ + d] * e;
      }
      int qg = qt * 16 + r;
      float val = (cacc[nt][i] + rel) / l_i[i];
      if (dim < DKC_)
        ctxc[((size_t)b * S_ + qg) * (H_ * DKC_) + h * DKC_ + dim] = f2bf(val);
      else
        ctxp[((size_t)b * S_ + qg) * (H_ * DKP_) + h * DKP_ + (dim - DKC_)] = f2bf(val);
    }
  }
}

// -------------------- host launch --------------------
extern "C" void kernel_launch(void* const* d_in, const int* in_sizes, int n_in,
                              void* d_out, int out_size, void* d_ws, size_t ws_size,
                              hipStream_t stream) {
  const float* key   = (const float*)d_in[0];
  const float* value = (const float*)d_in[1];
  const float* query = (const float*)d_in[2];
  const unsigned char* mask = (const unsigned char*)d_in[3]; // jnp bool_
  const float* Wq1 = (const float*)d_in[4];  const float* bq1 = (const float*)d_in[5];
  const float* Wk1 = (const float*)d_in[6];  const float* bk1 = (const float*)d_in[7];
  const float* Wv1 = (const float*)d_in[8];  const float* bv1 = (const float*)d_in[9];
  const float* Wq2 = (const float*)d_in[10]; const float* bq2 = (const float*)d_in[11];
  const float* Wk2 = (const float*)d_in[12]; const float* bk2 = (const float*)d_in[13];
  const float* Wv2 = (const float*)d_in[14]; const float* bv2 = (const float*)d_in[15];
  const float* Wo1 = (const float*)d_in[16]; const float* Wo2 = (const float*)d_in[17];
  const float* ek_c = (const float*)d_in[18]; const float* ek_p = (const float*)d_in[19];
  const float* ev_c = (const float*)d_in[20]; const float* ev_p = (const float*)d_in[21];
  (void)ws_size; (void)n_in; (void)in_sizes; (void)out_size;

  // ---- workspace carve-up ----
  char* base = (char*)d_ws;
  size_t off = 0;
  auto alloc = [&](size_t bytes) { char* p = base + off; off = (off + bytes + 255) & ~(size_t)255; return p; };
  const int M = B_ * S_;   // 8192 rows
  unsigned short* wq1t = (unsigned short*)alloc((size_t)DC_ * H_ * DKC_ * 2);
  unsigned short* wk1t = (unsigned short*)alloc((size_t)DC_ * H_ * DKC_ * 2);
  unsigned short* wv1t = (unsigned short*)alloc((size_t)DC_ * H_ * DKC_ * 2);
  unsigned short* wq2t = (unsigned short*)alloc((size_t)DP_ * H_ * DKP_ * 2);
  unsigned short* wk2t = (unsigned short*)alloc((size_t)DP_ * H_ * DKP_ * 2);
  unsigned short* wv2t = (unsigned short*)alloc((size_t)DP_ * H_ * DKP_ * 2);
  unsigned short* wo1t = (unsigned short*)alloc((size_t)H_ * DKC_ * DC_ * 2);
  unsigned short* wo2t = (unsigned short*)alloc((size_t)H_ * DKP_ * DP_ * 2);
  unsigned short* qbf = (unsigned short*)alloc((size_t)M * DC_ * 2);  // bf16 inputs
  unsigned short* kbf = (unsigned short*)alloc((size_t)M * DC_ * 2);
  unsigned short* vbf = (unsigned short*)alloc((size_t)M * DC_ * 2);
  unsigned short* Qb = (unsigned short*)alloc((size_t)B_ * H_ * S_ * DH_ * 2);
  unsigned short* Kb = (unsigned short*)alloc((size_t)B_ * H_ * S_ * DH_ * 2);
  unsigned short* Vt = (unsigned short*)alloc((size_t)B_ * H_ * S_ * DH_ * 2);
  unsigned short* Qc = (unsigned short*)alloc((size_t)M * H_ * DKC_ * 2);
  unsigned short* Kc = (unsigned short*)alloc((size_t)M * H_ * DKC_ * 2);
  unsigned short* Vc = (unsigned short*)alloc((size_t)M * H_ * DKC_ * 2);
  unsigned short* Qp = (unsigned short*)alloc((size_t)M * H_ * DKP_ * 2);
  unsigned short* Kp = (unsigned short*)alloc((size_t)M * H_ * DKP_ * 2);
  unsigned short* Vp = (unsigned short*)alloc((size_t)M * H_ * DKP_ * 2);
  // qbf/kbf/vbf are dead after build_heads -> reuse for R and ctx buffers.
  float*          Rr   = (float*)qbf;          // B*H*S*33 f32 = 8.7MB <= 12.6MB
  unsigned short* ctxc = kbf;                  // [M, 384] bf16
  unsigned short* ctxp = vbf;                  // [M, 128] bf16

  auto cvt = [&](const float* W, unsigned short* Wt, int K, int N) {
    int n = K * N;
    hipLaunchKernelGGL(convt_kernel, dim3((n + 255) / 256), dim3(256), 0, stream, W, Wt, K, N);
  };
  cvt(Wq1, wq1t, DC_, H_ * DKC_); cvt(Wk1, wk1t, DC_, H_ * DKC_); cvt(Wv1, wv1t, DC_, H_ * DKC_);
  cvt(Wq2, wq2t, DP_, H_ * DKP_); cvt(Wk2, wk2t, DP_, H_ * DKP_); cvt(Wv2, wv2t, DP_, H_ * DKP_);
  cvt(Wo1, wo1t, H_ * DKC_, DC_); cvt(Wo2, wo2t, H_ * DKP_, DP_);

  {
    int n = M * DC_;
    dim3 g((n + 255) / 256), blk(256);
    hipLaunchKernelGGL(cvt_input_kernel, g, blk, 0, stream, query, qbf, DC_);
    hipLaunchKernelGGL(cvt_input_kernel, g, blk, 0, stream, key,   kbf, DC_);
    hipLaunchKernelGGL(cvt_input_kernel, g, blk, 0, stream, value, vbf, DC_);
  }

  auto gemm_bf16 = [&](const unsigned short* A, int lda, const unsigned short* Wt,
                       const float* bias, unsigned short* out, int N, int K, float scale) {
    int tiles = (M / 16) * (N / 64);
    hipLaunchKernelGGL((wmma_gemm_bf16<false>), dim3((tiles + 3) / 4), dim3(128), 0, stream,
                       A, lda, Wt, bias, (void*)out, N, M / 16, N / 64, K, scale);
  };
  const float qscale = 0.125f; // 1/sqrt(64)
  gemm_bf16(qbf, DC_, wq1t, bq1, Qc, H_ * DKC_, DC_, qscale);
  gemm_bf16(kbf, DC_, wk1t, bk1, Kc, H_ * DKC_, DC_, 1.0f);
  gemm_bf16(vbf, DC_, wv1t, bv1, Vc, H_ * DKC_, DC_, 1.0f);
  gemm_bf16(qbf, DC_, wq2t, bq2, Qp, H_ * DKP_, DP_, qscale);  // first 256 cols of qbf
  gemm_bf16(kbf, DC_, wk2t, bk2, Kp, H_ * DKP_, DP_, 1.0f);
  gemm_bf16(vbf, DC_, wv2t, bv2, Vp, H_ * DKP_, DP_, 1.0f);

  {
    int n = B_ * H_ * S_ * DH_;
    dim3 g((n + 255) / 256), blk(256);
    hipLaunchKernelGGL((build_heads_kernel<false>), g, blk, 0, stream, Qc, Qp, Qb);
    hipLaunchKernelGGL((build_heads_kernel<false>), g, blk, 0, stream, Kc, Kp, Kb);
    hipLaunchKernelGGL((build_heads_kernel<true>),  g, blk, 0, stream, Vc, Vp, Vt);
  }
  {
    int n = B_ * H_ * S_ * ND_;
    hipLaunchKernelGGL(rel_logits_kernel, dim3((n + 255) / 256), dim3(256), 0, stream,
                       Qb, ek_c, ek_p, Rr);
  }
  hipLaunchKernelGGL(attn_kernel, dim3(B_ * H_ * (S_ / 16)), dim3(32), 0, stream,
                     Qb, Kb, Vt, Rr, mask, ev_c, ev_p, ctxc, ctxp);

  // output projections straight into d_out (f32, row stride 1024)
  {
    int N = DC_, K = H_ * DKC_;
    int tiles = (M / 16) * (N / 64);
    hipLaunchKernelGGL((wmma_gemm_bf16<true>), dim3((tiles + 3) / 4), dim3(128), 0, stream,
                       ctxc, K, wo1t, (const float*)nullptr, d_out, DM_, M / 16, N / 64, K, 1.0f);
  }
  {
    int N = DP_, K = H_ * DKP_;
    int tiles = (M / 16) * (N / 64);
    hipLaunchKernelGGL((wmma_gemm_bf16<true>), dim3((tiles + 3) / 4), dim3(128), 0, stream,
                       ctxp, K, wo2t, (const float*)nullptr, (float*)d_out + DC_, DM_,
                       M / 16, N / 64, K, 1.0f);
  }
}
